// STULayer_85323820303006
// MI455X (gfx1250) — matrix operand order repaired and checked by reading, not compiled, not run on hardware
//
#include <hip/hip_runtime.h>
#include <hip/hip_bf16.h>

typedef __attribute__((ext_vector_type(16))) __bf16 v16bf;
typedef __attribute__((ext_vector_type(8)))  __bf16 v8bf;
typedef __attribute__((ext_vector_type(2)))  __bf16 bf16x2;
typedef __attribute__((ext_vector_type(8)))  float  v8f;
typedef __attribute__((ext_vector_type(4)))  int    v4i;

#define B_     4
#define T_     2048
#define DIN    512
#define DOUT   512
#define NF     24
#define TT     32                 // t-tile
#define NTILES (T_/TT)            // 64
#define OT     128                // o-tile per block
#define OTILES (DOUT/OT)          // 4
#define ASTRIDE 536               // bf16 elems per LDS row: 16B aligned, 12-bank skew

// Workspace layout:
//   wsB : bf16, NF*DOUT*DIN elems, M pre-packed in WMMA B-fragment order (12.58 MB)
//   wsC : f32,  B_*NF*NTILES*DIN elems, per-tile partials -> exclusive carries (12.58 MB)

#if defined(__has_builtin)
#if __has_builtin(__builtin_amdgcn_global_load_async_to_lds_b128)
#define HAVE_ASYNC_LDS 1
#endif
#endif

typedef v4i __attribute__((address_space(1))) g_v4i;
typedef v4i __attribute__((address_space(3))) l_v4i;

__device__ __forceinline__ void async_copy_b128(const void* g, void* l) {
#ifdef HAVE_ASYNC_LDS
    // generic->AS1 via inttoptr; generic LDS ptr low 32 bits == LDS offset (ISA 10.2)
    g_v4i* gp = (g_v4i*)(unsigned long long)(size_t)g;
    l_v4i* lp = (l_v4i*)(unsigned)(size_t)l;
    __builtin_amdgcn_global_load_async_to_lds_b128(gp, lp, 0, 0);
#else
    *(float4*)l = *(const float4*)g;
#endif
}

__device__ __forceinline__ void async_wait0() {
#ifdef HAVE_ASYNC_LDS
    asm volatile("s_wait_asynccnt 0x0" ::: "memory");
#endif
}

// ---------------------------------------------------------------------------
// Kernel 0: pack M_weights (NF, DOUT, DIN) f32 into bf16 B-fragment layout.
// B fragment (32x16, K x N): lanes 0-15 hold K=0..15 (col N = lane),
// lanes 16-31 hold K=16..31; per-lane element e = K % 16 (16 contiguous bf16).
__global__ void pack_M_kernel(const float* __restrict__ M, __bf16* __restrict__ wsB) {
    int id   = blockIdx.x * 256 + threadIdx.x;   // 0 .. NF*2^18-1
    int e    = id & 15;
    int lane = (id >> 4) & 31;
    int ks   = (id >> 9) & 15;
    int ob   = (id >> 13) & 31;
    int n    = id >> 18;
    int o = ob * 16 + (lane & 15);
    int k = ks * 32 + (lane >> 4) * 16 + e;
    wsB[id] = (__bf16)M[((size_t)n * DOUT + o) * DIN + k];
}

// ---------------------------------------------------------------------------
// Kernel 1: per-tile partial sums  S[b,n,tile,f] = sum_{t in tile} phi[t,n]*x[b,t,f]
__global__ void partial_sums_kernel(const float* __restrict__ x,
                                    const float* __restrict__ filt,
                                    float* __restrict__ wsC) {
    __shared__ float phiT[TT][NF];
    int b     = blockIdx.x / NTILES;
    int tile  = blockIdx.x % NTILES;
    int tBase = tile * TT;
    int tid   = threadIdx.x;     // 256
    for (int i = tid; i < TT * NF; i += 256)
        phiT[i / NF][i % NF] = filt[(size_t)(tBase + i / NF) * NF + (i % NF)];
    __syncthreads();

    float acc0[NF], acc1[NF];
#pragma unroll
    for (int n = 0; n < NF; ++n) { acc0[n] = 0.f; acc1[n] = 0.f; }

    int f = tid * 2;
    const float* xb = x + ((size_t)b * T_ + tBase) * DIN;
    for (int t = 0; t < TT; ++t) {
        float2 xv = *(const float2*)&xb[(size_t)t * DIN + f];
#pragma unroll
        for (int n = 0; n < NF; ++n) {
            float p = phiT[t][n];
            acc0[n] += p * xv.x;
            acc1[n] += p * xv.y;
        }
    }
    for (int n = 0; n < NF; ++n) {
        size_t base = (((size_t)b * NF + n) * NTILES + tile) * DIN;
        float2 s; s.x = acc0[n]; s.y = acc1[n];
        *(float2*)&wsC[base + f] = s;
    }
}

// ---------------------------------------------------------------------------
// Kernel 2: sequential exclusive scan over tiles (64 steps), in place.
__global__ void scan_kernel(float* __restrict__ wsC) {
    int bn = blockIdx.x;          // b*NF + n
    int f  = threadIdx.x;         // 512
    float c = 0.f;
    size_t base = (size_t)bn * NTILES * DIN + f;
    for (int tile = 0; tile < NTILES; ++tile) {
        float s = wsC[base + (size_t)tile * DIN];
        wsC[base + (size_t)tile * DIN] = c;
        c += s;
    }
}

// ---------------------------------------------------------------------------
// Kernel 3: fused cumsum -> bf16 WMMA GEMM -> phi modulation -> out.
__global__ __launch_bounds__(256) void stu_main_kernel(
    const float* __restrict__ x, const float* __restrict__ filt,
    const __bf16* __restrict__ wsB, const float* __restrict__ wsC,
    float* __restrict__ out) {
    __shared__ float  xTile[TT][DIN];       // 65536 B, staged once via async copy
    __shared__ __bf16 aTile[TT][ASTRIDE];   // 34304 B
    __shared__ float  phiT[TT][NF];         //  3072 B

    int b     = blockIdx.x / NTILES;
    int tile  = blockIdx.x % NTILES;
    int oT    = blockIdx.y;                 // 0..3
    int tBase = tile * TT;
    int tid   = threadIdx.x;
    int w     = tid >> 5;                   // wave 0..7 (wave32)
    int lane  = tid & 31;
    int half  = lane >> 4;
    int lcol  = lane & 15;

    // Stage the contiguous 64KB x tile into LDS with async global->LDS copies.
    {
        const char* g = (const char*)(x + ((size_t)b * T_ + tBase) * DIN);
        char* l = (char*)&xTile[0][0];
#pragma unroll
        for (int i = 0; i < 16; ++i) {
            int off = (tid + i * 256) * 16;
            async_copy_b128(g + off, l + off);
        }
    }
    for (int i = tid; i < TT * NF; i += 256)
        phiT[i / NF][i % NF] = filt[(size_t)(tBase + i / NF) * NF + (i % NF)];
    async_wait0();
    __syncthreads();

    v8f outAcc[2];
#pragma unroll
    for (int ts = 0; ts < 2; ++ts)
#pragma unroll
        for (int r = 0; r < 8; ++r) outAcc[ts][r] = 0.f;

    int f = tid * 2;
    int ob = oT * 8 + w;

    for (int n = 0; n < NF; ++n) {
        // Phase A: f32 running cumsum seeded from carry; packed bf16x2 LDS stores.
        size_t cbase = (((size_t)b * NF + n) * NTILES + tile) * DIN;
        float2 c = *(const float2*)&wsC[cbase + f];
        for (int t = 0; t < TT; ++t) {
            float p = phiT[t][n];
            c.x += p * xTile[t][f];
            c.y += p * xTile[t][f + 1];
            bf16x2 pk;
            pk[0] = (__bf16)c.x;
            pk[1] = (__bf16)c.y;
            *(bf16x2*)&aTile[t][f] = pk;
        }
        __syncthreads();

        // Phase B: (32 x 512) @ (512 x 16 per wave) via v_wmma_f32_16x16x32_bf16
        v8f macc[2];
#pragma unroll
        for (int ts = 0; ts < 2; ++ts)
#pragma unroll
            for (int r = 0; r < 8; ++r) macc[ts][r] = 0.f;

        const __bf16* bptr = wsB + (((size_t)n * 32 + ob) * 16 * 32 + lane) * 16;
#pragma unroll
        for (int ks = 0; ks < 16; ++ks) {
            v16bf bfrag = *(const v16bf*)(bptr + (size_t)ks * 512);
#pragma unroll
            for (int ts = 0; ts < 2; ++ts) {
                // A fragment: lane half h reads K = ks*32 + 16h..16h+15
                const __bf16* ap = &aTile[ts * 16 + lcol][ks * 32 + half * 16];
                v8bf lo = *(const v8bf*)ap;
                v8bf hi = *(const v8bf*)(ap + 8);
                v16bf afrag = __builtin_shufflevector(
                    lo, hi, 0,1,2,3,4,5,6,7,8,9,10,11,12,13,14,15);
                macc[ts] = __builtin_amdgcn_wmma_f32_16x16x32_bf16(
                    false, afrag, false, bfrag, (short)0, macc[ts], false, false);
            }
        }

        // Phase C: scale rows by phi[t,n] (f32) and accumulate.
#pragma unroll
        for (int ts = 0; ts < 2; ++ts)
#pragma unroll
            for (int r = 0; r < 8; ++r) {
                int row = ts * 16 + r + 8 * half;   // C/D layout: M = r + 8*half
                outAcc[ts][r] += phiT[row][n] * macc[ts][r];
            }
        __syncthreads();   // aTile consumed before next n overwrites it
    }

    int oIdx = oT * OT + w * 16 + lcol;
#pragma unroll
    for (int ts = 0; ts < 2; ++ts)
#pragma unroll
        for (int r = 0; r < 8; ++r) {
            int t = tBase + ts * 16 + r + 8 * half;
            out[((size_t)b * T_ + t) * DOUT + oIdx] = outAcc[ts][r];
        }
}

// ---------------------------------------------------------------------------
extern "C" void kernel_launch(void* const* d_in, const int* in_sizes, int n_in,
                              void* d_out, int out_size, void* d_ws, size_t ws_size,
                              hipStream_t stream) {
    const float* x    = (const float*)d_in[0];   // (4,2048,512) f32
    const float* filt = (const float*)d_in[1];   // (2048,24)   f32
    const float* M    = (const float*)d_in[2];   // (24,512,512) f32
    float* out = (float*)d_out;                  // (4,2048,512) f32

    __bf16* wsB = (__bf16*)d_ws;                                   // 12.58 MB
    float*  wsC = (float*)((char*)d_ws +
                           (size_t)NF * DOUT * DIN * sizeof(__bf16)); // 12.58 MB

    pack_M_kernel<<<(NF * DOUT * DIN) / 256, 256, 0, stream>>>(M, wsB);
    partial_sums_kernel<<<B_ * NTILES, 256, 0, stream>>>(x, filt, wsC);
    scan_kernel<<<B_ * NF, 512, 0, stream>>>(wsC);
    dim3 grid(B_ * NTILES, OTILES);
    stu_main_kernel<<<grid, 256, 0, stream>>>(x, filt, wsB, wsC, out);
}
